// PEGCN_12704513262276
// MI455X (gfx1250) — matrix operand order repaired and checked by reading, not compiled, not run on hardware
//
#include <hip/hip_runtime.h>
#include <hip/hip_bf16.h>
#include <math.h>

// ---- problem constants (from reference) ----
#define NN   16384   // nodes
#define KNN  16      // neighbors
#define FC   2       // coord features
#define FX   32      // attr features
#define EE   256     // embedding
#define EE2  512     // 2*E

// ---- CDNA5 WMMA types (wave32; probe-confirmed builtin) ----
typedef __attribute__((ext_vector_type(16))) __bf16 v16bf;
typedef __attribute__((ext_vector_type(8)))  float  v8f;

union Frag16 { v16bf v; uint4 q[2]; };

// ============================================================
// KNN: one thread per node, candidates tiled through LDS.
// Produces src indices (dst-major), edge weights, and dinv.
// ============================================================
__global__ void knn_kernel(const float* __restrict__ c,
                           int*   __restrict__ srcIdx,
                           float* __restrict__ ew,
                           float* __restrict__ dinv) {
  __shared__ float2 s_c[256];
  const int i = blockIdx.x * 256 + threadIdx.x;
  const float cix = c[i * 2 + 0];
  const float ciy = c[i * 2 + 1];

  float bestd[KNN];
  int   besti[KNN];
  #pragma unroll
  for (int k = 0; k < KNN; ++k) { bestd[k] = 3.0e38f; besti[k] = 0; }

  for (int tile = 0; tile < NN; tile += 256) {
    __syncthreads();
    s_c[threadIdx.x] = *reinterpret_cast<const float2*>(&c[(tile + threadIdx.x) * 2]);
    __syncthreads();
    for (int jj = 0; jj < 256; ++jj) {
      const int cand = tile + jj;
      if (cand == i) continue;                       // diagonal = +inf in reference
      const float dx = cix - s_c[jj].x;
      const float dy = ciy - s_c[jj].y;
      const float d2 = dx * dx + dy * dy;
      if (d2 < bestd[KNN - 1]) {                     // insertion into sorted top-16
        int p = KNN - 1;
        while (p > 0 && bestd[p - 1] > d2) {
          bestd[p] = bestd[p - 1];
          besti[p] = besti[p - 1];
          --p;
        }
        bestd[p] = d2;
        besti[p] = cand;
      }
    }
  }

  // edge weights + degree (dst-major edges: deg[i] = sum_k ew + selfloop 1)
  float deg = 1.0f;
  #pragma unroll
  for (int k = 0; k < KNN; ++k) {
    float w = __expf(-0.5f * sqrtf(bestd[k]));
    if (w > 10.0f) w = 0.0f;                         // reference clamp (never fires)
    srcIdx[i * KNN + k] = besti[k];
    ew[i * KNN + k] = w;
    deg += w;
  }
  dinv[i] = rsqrtf(deg);
}

// ============================================================
// Pack W (Kd x Ed f32, row major) into fragment-major bf16:
//   Wpk[kblk][col][kk] , kblk = k>>5, kk = k&31
// so one lane's 16-element B fragment (K = k0 + hi*16 + {0..15},
// fixed col) is 32 contiguous bytes. Zero-pads K up to Kpad.
// ============================================================
__global__ void pack_w_bf16(const float* __restrict__ W,
                            __bf16* __restrict__ Wpk,
                            int Kd, int Ed) {
  const int idx = blockIdx.x * 256 + threadIdx.x;    // < Kpad * Ed
  const int k   = idx >> 8 << 8 ? 0 : 0;             // (placeholder removed below)
  (void)k;
  // idx -> (k, col) with k-major over Kpad
  const int kk_total = idx / Ed;                     // k in [0, Kpad)
  const int col      = idx - kk_total * Ed;
  const float v = (kk_total < Kd) ? W[kk_total * Ed + col] : 0.0f;
  const int kblk = kk_total >> 5;
  const int kk   = kk_total & 31;
  Wpk[((size_t)kblk * Ed + col) * 32 + kk] = (__bf16)v;
}

// ============================================================
// WMMA GEMM:  out = [res +] maybe_relu(A @ W + bias)
//   A: N x Kd (f32, row major)  -> bf16 tile staged in LDS
//   Wpk: packed bf16 (see pack_w_bf16), already zero-padded
//   block = 256 thr = 8 wave32; block tile = 16 rows x 128 cols
//   inner loop: 2x ds_load_b128 + 2x global_load_b128 + 1x wmma
// ============================================================
__global__ void wmma_gemm_bias_act(const float* __restrict__ A,
                                   const __bf16* __restrict__ Wpk,
                                   const float* __restrict__ bias,
                                   const float* __restrict__ res,
                                   float* __restrict__ out,
                                   int Kd, int Kpad, int Ed, int do_relu) {
  extern __shared__ char smem[];
  __bf16* lds_a = reinterpret_cast<__bf16*>(smem);   // [16][Kpad] bf16

  const int rowBase = blockIdx.x * 16;
  const int tid = threadIdx.x;

  // Stage + convert A tile (16*Kpad is a multiple of 256 -> no divergence).
  const int tot = 16 * Kpad;
  for (int idx = tid; idx < tot; idx += 256) {
    const int r = idx / Kpad;
    const int k = idx - r * Kpad;
    const float v = (k < Kd) ? A[(rowBase + r) * Kd + k] : 0.0f;
    lds_a[idx] = (__bf16)v;
  }
  __syncthreads();

  const int lane = tid & 31;
  const int wave = tid >> 5;
  const int m    = lane & 15;           // A row within tile / B column lane
  const int hi   = lane >> 4;           // 0: lanes 0-15, 1: lanes 16-31
  const int col  = blockIdx.y * 128 + wave * 16 + m;

  const __bf16* arow = &lds_a[m * Kpad + hi * 16];
  const __bf16* bp   = Wpk + (size_t)col * 32 + hi * 16;
  const size_t  bstep = (size_t)Ed * 32;             // one 32-K block of packed W

  v8f acc = {};
  for (int k0 = 0; k0 < Kpad; k0 += 32) {
    // A fragment (contiguous K order, matching packed B):
    //   lanes 0-15:  K = k0+{0..15}
    //   lanes 16-31: K = k0+{16..31}
    Frag16 af;
    af.q[0] = *reinterpret_cast<const uint4*>(arow + k0);        // ds_load_b128
    af.q[1] = *reinterpret_cast<const uint4*>(arow + k0 + 8);    // ds_load_b128

    // B fragment: 32 contiguous bytes of packed W (pre-padded, no guards)
    Frag16 bf;
    bf.q[0] = *reinterpret_cast<const uint4*>(bp);               // global_load_b128
    bf.q[1] = *reinterpret_cast<const uint4*>(bp + 8);           // global_load_b128
    bp += bstep;

    // D = A x B + C, f32 accumulate (v_wmma_f32_16x16x32_bf16)
    acc = __builtin_amdgcn_wmma_f32_16x16x32_bf16(
        /*neg_a=*/false, af.v, /*neg_b=*/false, bf.v,
        /*c_mod=*/(short)0, acc, /*reuse_a=*/false, /*reuse_b=*/false);
  }

  // C/D layout: VGPR v -> M = v (lanes 0-15) or v+8 (lanes 16-31), N = lane&15
  #pragma unroll
  for (int v = 0; v < 8; ++v) {
    const int row = rowBase + v + hi * 8;
    float val = acc[v];
    if (bias)    val += bias[col];
    if (do_relu) val  = fmaxf(val, 0.0f);
    if (res)     val += res[row * Ed + col];
    out[row * Ed + col] = val;
  }
}

// ============================================================
// GCN aggregation (gather form, no atomics):
//   out[i,:] = res[i,:] + relu( dinv[i]^2*hw[i,:]
//              + dinv[i]*sum_k ew_k*dinv[src_k]*hw[src_k,:] + b )
// ============================================================
__global__ void gcn_aggregate(const float* __restrict__ hw,
                              const int*   __restrict__ srcIdx,
                              const float* __restrict__ ew,
                              const float* __restrict__ dinv,
                              const float* __restrict__ bias,
                              const float* __restrict__ res,
                              float* __restrict__ out) {
  __shared__ int   s_src[KNN];
  __shared__ float s_w[KNN];
  const int i = blockIdx.x;
  const int j = threadIdx.x;            // 0..255 (one column each)
  const float di = dinv[i];
  if (j < KNN) {
    const int s = srcIdx[i * KNN + j];
    s_src[j] = s;
    s_w[j]   = ew[i * KNN + j] * dinv[s] * di;
  }
  __syncthreads();

  float acc = di * di * hw[i * EE + j];  // self loop: dinv[i]*1*dinv[i]
  #pragma unroll
  for (int k = 0; k < KNN; ++k)
    acc += s_w[k] * hw[s_src[k] * EE + j];
  acc += bias[j];
  acc = fmaxf(acc, 0.0f);
  acc += res[i * EE + j];
  out[i * EE + j] = acc;
}

// ============================================================
// combined = [g3 | m3]; also emit g3/m3 output regions.
// d_out layout: mean[N], sigma[N], g3[N*E], m3[N*E]
// ============================================================
__global__ void combine_kernel(const float* __restrict__ g3,
                               const float* __restrict__ m3,
                               float* __restrict__ combined,
                               float* __restrict__ dout) {
  const int idx = blockIdx.x * 256 + threadIdx.x;   // < N*512
  const int i = idx >> 9;
  const int j = idx & 511;
  float v;
  if (j < EE) {
    v = g3[i * EE + j];
    dout[2 * NN + i * EE + j] = v;
  } else {
    v = m3[i * EE + (j - EE)];
    dout[2 * NN + NN * EE + i * EE + (j - EE)] = v;
  }
  combined[idx] = v;
}

// ============================================================
// Head: out2 = hidden @ W_d2 + b_d2 ; mean / sigma transform
// ============================================================
__global__ void head_kernel(const float* __restrict__ hidden,
                            const float* __restrict__ W2,
                            const float* __restrict__ b2,
                            float* __restrict__ outMean,
                            float* __restrict__ outSigma) {
  const int i = blockIdx.x * 256 + threadIdx.x;
  float a0 = b2[0], a1 = b2[1];
  for (int k = 0; k < EE2; ++k) {
    const float h = hidden[i * EE2 + k];
    a0 += h * W2[k * 2 + 0];
    a1 += h * W2[k * 2 + 1];
  }
  outMean[i] = a0;
  const float sp = (a1 > 20.0f) ? a1 : log1pf(__expf(a1));
  outSigma[i] = 0.2f + 0.8f * sp;
}

// ============================================================
// Host-side orchestration
// ============================================================
extern "C" void kernel_launch(void* const* d_in, const int* in_sizes, int n_in,
                              void* d_out, int out_size, void* d_ws, size_t ws_size,
                              hipStream_t stream) {
  const float* c        = (const float*)d_in[0];
  const float* x        = (const float*)d_in[1];
  const float* W_coords = (const float*)d_in[2];
  const float* b_coords = (const float*)d_in[3];
  const float* W_attri  = (const float*)d_in[4];
  const float* b_attri  = (const float*)d_in[5];
  const float* W_g1 = (const float*)d_in[6];   const float* b_g1 = (const float*)d_in[7];
  const float* W_g2 = (const float*)d_in[8];   const float* b_g2 = (const float*)d_in[9];
  const float* W_g3 = (const float*)d_in[10];  const float* b_g3 = (const float*)d_in[11];
  const float* W_f1 = (const float*)d_in[12];  const float* b_f1 = (const float*)d_in[13];
  const float* W_f2 = (const float*)d_in[14];  const float* b_f2 = (const float*)d_in[15];
  const float* W_f3 = (const float*)d_in[16];  const float* b_f3 = (const float*)d_in[17];
  const float* W_d1 = (const float*)d_in[18];  const float* b_d1 = (const float*)d_in[19];
  const float* W_d2 = (const float*)d_in[20];  const float* b_d2 = (const float*)d_in[21];

  // bump allocator over workspace (float units)
  float* ws = (float*)d_ws;
  size_t off = 0;
  auto alloc = [&](size_t elems) { float* p = ws + off; off += (elems + 7) & ~(size_t)7; return p; };

  int*   srcIdx   = (int*)alloc((size_t)NN * KNN);
  float* ew       = alloc((size_t)NN * KNN);
  float* dinv     = alloc(NN);
  float* coords0  = alloc((size_t)NN * EE);
  float* attri0   = alloc((size_t)NN * EE);
  float* gcur     = alloc((size_t)NN * EE);
  float* hw       = alloc((size_t)NN * EE);
  float* m_a      = alloc((size_t)NN * EE);
  float* m_b      = alloc((size_t)NN * EE);
  float* combined = alloc((size_t)NN * EE2);
  float* hidden   = alloc((size_t)NN * EE2);
  (void)ws_size; (void)in_sizes; (void)n_in; (void)out_size;

  // pack-then-gemm: packed W buffers come from the bump allocator so the
  // sequence is identical every call (graph-capture safe, deterministic).
  auto gemm = [&](const float* A, const float* W, const float* bias,
                  const float* res, float* out, int Kd, int Ed, int relu) {
    const int Kpad = (Kd + 31) & ~31;
    __bf16* Wpk = (__bf16*)alloc(((size_t)Kpad * Ed) / 2);   // bf16 elems -> float units
    pack_w_bf16<<<(Kpad * Ed) / 256, 256, 0, stream>>>(W, Wpk, Kd, Ed);
    dim3 grid(NN / 16, Ed / 128);
    const size_t shmem = (size_t)16 * Kpad * sizeof(__bf16);
    wmma_gemm_bias_act<<<grid, dim3(256), shmem, stream>>>(
        A, Wpk, bias, res, out, Kd, Kpad, Ed, relu);
  };

  float* outf = (float*)d_out;

  // graph + edge weights + normalization
  knn_kernel<<<NN / 256, 256, 0, stream>>>(c, srcIdx, ew, dinv);

  // projections
  gemm(c, W_coords, b_coords, nullptr, coords0, FC, EE, 0);
  gemm(x, W_attri,  b_attri,  nullptr, attri0,  FX, EE, 0);

  // layer 1
  gemm(c, W_g1, nullptr, nullptr, hw, FC, EE, 0);
  gcn_aggregate<<<NN, EE, 0, stream>>>(hw, srcIdx, ew, dinv, b_g1, coords0, gcur);
  gemm(x, W_f1, b_f1, attri0, m_a, FX, EE, 1);

  // layer 2
  gemm(gcur, W_g2, nullptr, nullptr, hw, EE, EE, 0);
  gcn_aggregate<<<NN, EE, 0, stream>>>(hw, srcIdx, ew, dinv, b_g2, gcur, gcur);
  gemm(m_a, W_f2, b_f2, m_a, m_b, EE, EE, 1);

  // layer 3
  gemm(gcur, W_g3, nullptr, nullptr, hw, EE, EE, 0);
  gcn_aggregate<<<NN, EE, 0, stream>>>(hw, srcIdx, ew, dinv, b_g3, gcur, gcur);
  gemm(m_b, W_f3, b_f3, m_b, m_a, EE, EE, 1);

  // concat (also writes g3/m3 outputs), dense 512->512, head
  combine_kernel<<<(NN * EE2) / 256, 256, 0, stream>>>(gcur, m_a, combined, outf);
  gemm(combined, W_d1, b_d1, nullptr, hidden, EE2, EE2, 1);
  head_kernel<<<NN / 256, 256, 0, stream>>>(hidden, W_d2, b_d2, outf, outf + NN);
}
